// MultiHeadLatentAttention_55843164782737
// MI455X (gfx1250) — matrix operand order repaired and checked by reading, not compiled, not run on hardware
//
#include <hip/hip_runtime.h>

// ---------------------------------------------------------------------------
// MLA attention for MI455X (gfx1250).
// Compute-bound (~163 GFLOP, ~100MB working set -> fully L2-resident), so all
// matrix math runs on v_wmma_f32_16x16x32_bf16 (f32 accumulate).
// GEMMs: 128x128 block tile per 8-wave workgroup, double-buffered LDS staging
// filled with GLOBAL_LOAD_ASYNC_TO_LDS_B128 (ASYNCcnt) when the toolchain
// exposes the builtin; each wave computes a 64x32 sub-tile (8 WMMAs / k-step).
// Attention is computed transposed (S^T = K*Q^T, O^T = V^T*P^T) so softmax
// rows live in lanes (one shfl_xor(16) per reduction).
// ---------------------------------------------------------------------------

#define BB   2
#define SS   2048
#define DD   2048
#define HH   16
#define HDIM 128
#define LATD 512
#define MTOT (BB * SS)   // 4096 token rows

typedef __bf16 bf16_t;
typedef __attribute__((ext_vector_type(8)))  __bf16 v8bf;
typedef __attribute__((ext_vector_type(16))) __bf16 v16bf;
typedef __attribute__((ext_vector_type(8)))  float  v8f;
typedef __attribute__((ext_vector_type(4)))  int    v4i;
// address-space qualified element types for the async-to-LDS builtin
typedef __attribute__((address_space(1))) v4i gv4i;   // global
typedef __attribute__((address_space(3))) v4i lv4i;   // LDS

#if __has_builtin(__builtin_amdgcn_global_load_async_to_lds_b128)
#define ASYNC_LDS 1
#else
#define ASYNC_LDS 0
#endif

static __device__ __forceinline__ bf16_t f2bf(float f) {
  // round-to-nearest-even fp32 -> bf16
  unsigned u = __builtin_bit_cast(unsigned, f);
  unsigned r = u + 0x7FFFu + ((u >> 16) & 1u);
  unsigned short h = (unsigned short)(r >> 16);
  return __builtin_bit_cast(bf16_t, h);
}

static __device__ __forceinline__ v16bf cat8(v8bf lo, v8bf hi) {
  return __builtin_shufflevector(lo, hi, 0,1,2,3,4,5,6,7,8,9,10,11,12,13,14,15);
}

// 16-byte global -> LDS copy: async (CDNA5 ASYNCcnt path) or sync fallback
static __device__ __forceinline__ void cp16(bf16_t* lds, const bf16_t* g) {
#if ASYNC_LDS
  __builtin_amdgcn_global_load_async_to_lds_b128((gv4i*)g, (lv4i*)lds, 0, 0);
#else
  *(v8bf*)lds = *(const v8bf*)g;
#endif
}

static __device__ __forceinline__ void async_wait() {
#if ASYNC_LDS
#if __has_builtin(__builtin_amdgcn_s_wait_asynccnt)
  __builtin_amdgcn_s_wait_asynccnt(0);
#else
  asm volatile("s_wait_asynccnt 0x0" ::: "memory");
#endif
#endif
}

// ---------------------------------------------------------------------------
// elementwise fp32 -> bf16
// ---------------------------------------------------------------------------
__global__ __launch_bounds__(256) void cvt_bf16(const float* __restrict__ in,
                                                bf16_t* __restrict__ out, int n) {
  int i = blockIdx.x * 256 + threadIdx.x;
  if (i < n) out[i] = f2bf(in[i]);
}

// ---------------------------------------------------------------------------
// W[K][N] fp32 -> Wt[N][K] bf16 (LDS-tiled 32x32)
// ---------------------------------------------------------------------------
__global__ __launch_bounds__(256) void transpose_cvt(const float* __restrict__ W,
                                                     bf16_t* __restrict__ Wt,
                                                     int K, int N) {
  __shared__ float tile[32][33];
  const int kb = blockIdx.x * 32, nb = blockIdx.y * 32;
  const int tx = threadIdx.x & 31, ty = threadIdx.x >> 5;
  for (int r = ty; r < 32; r += 8)
    tile[r][tx] = W[(size_t)(kb + r) * N + nb + tx];
  __syncthreads();
  for (int r = ty; r < 32; r += 8)
    Wt[(size_t)(nb + r) * K + kb + tx] = f2bf(tile[tx][r]);
}

// ---------------------------------------------------------------------------
// RoPE on fp32 tensor [B,S,H,HD] (interleaved pairs), output bf16 same shape
// ---------------------------------------------------------------------------
__global__ __launch_bounds__(256) void rope_cvt(const float* __restrict__ T,
                                                const float* __restrict__ freqs,
                                                bf16_t* __restrict__ O) {
  size_t i = (size_t)blockIdx.x * 256 + threadIdx.x;     // pair index
  int p = (int)(i & (HDIM / 2 - 1));                     // 0..63
  size_t rest = i >> 6;                                  // (b*S+s)*H + h
  int s = (int)((rest >> 4) & (SS - 1));
  float f  = freqs[s * (HDIM / 2) + p];
  float cs = __cosf(f), sn = __sinf(f);
  float t0 = T[2 * i], t1 = T[2 * i + 1];
  O[2 * i]     = f2bf(t0 * cs - t1 * sn);
  O[2 * i + 1] = f2bf(t0 * sn + t1 * cs);
}

// ---------------------------------------------------------------------------
// V fp32 [B,S,H,HD] -> Vt bf16 [B,H,HD,S]
// ---------------------------------------------------------------------------
__global__ __launch_bounds__(256) void v_transpose(const float* __restrict__ V,
                                                   bf16_t* __restrict__ Vt) {
  size_t i = (size_t)blockIdx.x * 256 + threadIdx.x;     // output index
  int s  = (int)(i & (SS - 1));
  size_t r = i >> 11;
  int hd = (int)(r & (HDIM - 1));
  size_t r2 = r >> 7;
  int h  = (int)(r2 & (HH - 1));
  int b  = (int)(r2 >> 4);
  size_t src = (((size_t)b * SS + s) * HH + h) * HDIM + hd;
  Vt[i] = f2bf(V[src]);
}

// ---------------------------------------------------------------------------
// C[M][N] fp32 = A[M][K] bf16 (row-major) x Bt[N][K] bf16 (B transposed).
// Block = 8 waves -> 128x128 output tile; per 32-wide k-step both 128x32
// tiles are staged in double-buffered LDS (rows padded to 40 bf16 = 80B so
// the 16 row-strided b128 fragment reads are bank-conflict-free).
// Wave (wm,wn) in a 2x4 grid computes a 64x32 sub-tile: 4 A-frags x 2 B-frags
// = 8 WMMAs per k-step. Fragment layouts per CDNA5 ISA 7.12.2.
// ---------------------------------------------------------------------------
#define LROW 40   // padded LDS row length in bf16 elements (32 data + 8 pad)

__global__ __launch_bounds__(256) void gemm_bt(const bf16_t* __restrict__ A,
                                               const bf16_t* __restrict__ Bt,
                                               float* __restrict__ C,
                                               int M, int N, int K) {
  __shared__ bf16_t lA[2][128 * LROW];
  __shared__ bf16_t lB[2][128 * LROW];

  const int tid  = threadIdx.x;
  const int lane = tid & 31;
  const int w    = tid >> 5;
  const int wm = w >> 2, wn = w & 3;           // 2x4 wave grid
  const int li = lane & 15, half = lane >> 4;
  const int m0 = blockIdx.y * 128;
  const int n0 = blockIdx.x * 128;

  const bf16_t* Abase = A  + (size_t)m0 * K;
  const bf16_t* Bbase = Bt + (size_t)n0 * K;

  // cooperative 128x32 tile copy: 512 x 16B chunks over 256 threads
  const int crow = tid >> 1;                    // rows 0..127 (2 threads/row)
  const int cseg2 = (tid & 1) * 2;              // segments {0,1} or {2,3}
  auto issue_copy = [&](int buf, int k0) {
#pragma unroll
    for (int i = 0; i < 2; ++i) {
      const int seg = cseg2 + i;                // 0..3, 8 bf16 each
      cp16(&lA[buf][crow * LROW + seg * 8], Abase + (size_t)crow * K + k0 + seg * 8);
      cp16(&lB[buf][crow * LROW + seg * 8], Bbase + (size_t)crow * K + k0 + seg * 8);
    }
  };

  v8f acc[4][2] = {};

  const int KT = K >> 5;
  int cur = 0;
  issue_copy(0, 0);
  for (int kt = 0; kt < KT; ++kt) {
    async_wait();
    __syncthreads();
    if (kt + 1 < KT) issue_copy(cur ^ 1, (kt + 1) * 32);

    // A fragments: 4 m-subtiles of this wave
    v16bf af[4];
#pragma unroll
    for (int t = 0; t < 4; ++t) {
      const bf16_t* p = &lA[cur][(64 * wm + 16 * t + li) * LROW + half * 8];
      af[t] = cat8(*(const v8bf*)p, *(const v8bf*)(p + 16));
    }
#pragma unroll
    for (int j = 0; j < 2; ++j) {
      const bf16_t* p = &lB[cur][(32 * wn + 16 * j + li) * LROW + half * 8];
      v16bf bfr = cat8(*(const v8bf*)p, *(const v8bf*)(p + 16));
#pragma unroll
      for (int t = 0; t < 4; ++t)
        acc[t][j] = __builtin_amdgcn_wmma_f32_16x16x32_bf16(
            false, af[t], false, bfr, (short)0, acc[t][j], false, false);
    }
    __syncthreads();
    cur ^= 1;
  }

  // store: C-tile lane holds col (lane&15), rows (lane>>4)*8 + v
#pragma unroll
  for (int t = 0; t < 4; ++t) {
    const int mbase = m0 + 64 * wm + 16 * t + half * 8;
#pragma unroll
    for (int j = 0; j < 2; ++j) {
      const int n = n0 + 32 * wn + 16 * j + li;
#pragma unroll
      for (int v = 0; v < 8; ++v)
        C[(size_t)(mbase + v) * N + n] = acc[t][j][v];
    }
  }
}

// ---------------------------------------------------------------------------
// Flash attention (transposed): one wave per (b, h, 16-query tile).
//   S^T = K_chunk(16k x 32hd WMMAs) * Q^T   -> lane = query column
//   online softmax per query: in-lane reduce over 16 keys + shfl_xor(16)
//   O^T(128hd x 16q) += V^T * P^T           -> 8 WMMA C-tiles
// Qb,Kb: bf16 [B,S,H,HD]; Vt: bf16 [B,H,HD,S]; Ob: bf16 [B*S, H*HD]
// ---------------------------------------------------------------------------
__global__ __launch_bounds__(256) void mla_attn(const bf16_t* __restrict__ Qb,
                                                const bf16_t* __restrict__ Kb,
                                                const bf16_t* __restrict__ Vt,
                                                bf16_t* __restrict__ Ob) {
  const int lane = threadIdx.x & 31;
  const int wid  = blockIdx.x * 8 + (threadIdx.x >> 5);
  const int qt = wid & (SS / 16 - 1);
  const int h  = (wid >> 7) & (HH - 1);
  const int b  = wid >> 11;
  const int li   = lane & 15;
  const int half = lane >> 4;
  const int qq   = qt * 16 + li;              // this lane's query index

  const size_t bh_qk = (size_t)b * SS * (HH * HDIM) + (size_t)h * HDIM;
  const bf16_t* Qrow = Qb + bh_qk + (size_t)qq * (HH * HDIM);

  // Q^T B-fragments, one per 32-wide hd chunk, reused for all key chunks
  v16bf qfrag[4];
#pragma unroll
  for (int c = 0; c < 4; ++c) {
    const bf16_t* p = Qrow + c * 32 + half * 8;
    qfrag[c] = cat8(*(const v8bf*)p, *(const v8bf*)(p + 16));
  }

  v8f o[8] = {};                               // O^T accumulators (8 hd tiles)
  float m_i = -1e30f, l_i = 0.0f;
  const float scale = 0.08838834764831845f;    // 1/sqrt(128)
  const bf16_t* Vbase = Vt + ((size_t)b * HH + h) * (size_t)HDIM * SS;

  const int nch = (qt + 2) >> 1;               // 32-key chunks needed (causal)
  for (int ch = 0; ch < nch; ++ch) {
    const int kb0 = ch * 32;
    const bf16_t* Krow0 = Kb + bh_qk + (size_t)(kb0 + li) * (HH * HDIM);
    const bf16_t* Krow1 = Krow0 + (size_t)16 * (HH * HDIM);

    v8f st0 = {}, st1 = {};                    // S^T tiles: keys 0..15 / 16..31
#pragma unroll
    for (int c = 0; c < 4; ++c) {
      const bf16_t* kp0 = Krow0 + c * 32 + half * 8;
      v16bf kf0 = cat8(*(const v8bf*)kp0, *(const v8bf*)(kp0 + 16));
      st0 = __builtin_amdgcn_wmma_f32_16x16x32_bf16(
          false, kf0, false, qfrag[c], (short)0, st0, false, false);
      const bf16_t* kp1 = Krow1 + c * 32 + half * 8;
      v16bf kf1 = cat8(*(const v8bf*)kp1, *(const v8bf*)(kp1 + 16));
      st1 = __builtin_amdgcn_wmma_f32_16x16x32_bf16(
          false, kf1, false, qfrag[c], (short)0, st1, false, false);
    }

    // scale + causal mask + chunk max (per query column)
    float p0[8], p1[8];
    float mx = -1e30f;
#pragma unroll
    for (int v = 0; v < 8; ++v) {
      const int k0 = kb0 + half * 8 + v;
      const int k1 = k0 + 16;
      float s0 = (k0 <= qq) ? st0[v] * scale : -1e30f;
      float s1 = (k1 <= qq) ? st1[v] * scale : -1e30f;
      p0[v] = s0; p1[v] = s1;
      mx = fmaxf(mx, fmaxf(s0, s1));
    }
    mx = fmaxf(mx, __shfl_xor(mx, 16, 32));
    const float m_new = fmaxf(m_i, mx);
    const float corr  = __expf(m_i - m_new);

    float lsum = 0.0f;
#pragma unroll
    for (int v = 0; v < 8; ++v) {
      p0[v] = __expf(p0[v] - m_new);
      p1[v] = __expf(p1[v] - m_new);
      lsum += p0[v] + p1[v];
    }
    lsum += __shfl_xor(lsum, 16, 32);
    l_i = l_i * corr + lsum;
    m_i = m_new;

    // build P^T (32k x 16q) B-fragment: one cross-half exchange
    v16bf pfrag;
#pragma unroll
    for (int v = 0; v < 8; ++v) {
      float o0 = __shfl_xor(p0[v], 16, 32);
      float o1 = __shfl_xor(p1[v], 16, 32);
      pfrag[v]     = f2bf(half ? o1 : p0[v]);   // k = 16*half + v of chunk
      pfrag[v + 8] = f2bf(half ? p1[v] : o0);   // k = 16*half + 8 + v of chunk
    }

    // O^T = corr*O^T + V^T * P^T over 8 hd tiles
#pragma unroll
    for (int c = 0; c < 8; ++c) {
#pragma unroll
      for (int v = 0; v < 8; ++v) o[c][v] *= corr;
      const bf16_t* vp = Vbase + (size_t)(c * 16 + li) * SS + kb0 + half * 16;
      v16bf vfrag = cat8(*(const v8bf*)vp, *(const v8bf*)(vp + 8));
      o[c] = __builtin_amdgcn_wmma_f32_16x16x32_bf16(
          false, vfrag, false, pfrag, (short)0, o[c], false, false);
    }
  }

  // normalize and store: row hd = c*16 + half*8 + v, col q = qq
  const float inv_l = 1.0f / l_i;
  bf16_t* orow = Ob + (size_t)(b * SS + qq) * DD + h * HDIM;
#pragma unroll
  for (int c = 0; c < 8; ++c) {
    v8bf pk;
#pragma unroll
    for (int v = 0; v < 8; ++v) pk[v] = f2bf(o[c][v] * inv_l);
    *reinterpret_cast<v8bf*>(orow + c * 16 + half * 8) = pk;
  }
}

// ---------------------------------------------------------------------------
// launch: x -> bf16; weights -> bf16 transposed; Q/KV GEMMs; RoPE; V^T;
// flash attention; output projection.  Workspace peak ~178 MB.
// ---------------------------------------------------------------------------
extern "C" void kernel_launch(void* const* d_in, const int* in_sizes, int n_in,
                              void* d_out, int out_size, void* d_ws, size_t ws_size,
                              hipStream_t stream) {
  (void)in_sizes; (void)n_in; (void)out_size; (void)ws_size;
  const float* x     = (const float*)d_in[0];
  const float* freqs = (const float*)d_in[1];
  /* d_in[2] = additive mask: causal, computed analytically in-kernel */
  const float* Wq    = (const float*)d_in[3];
  const float* Wdown = (const float*)d_in[4];
  const float* Wkup  = (const float*)d_in[5];
  const float* Wvup  = (const float*)d_in[6];
  const float* Wo    = (const float*)d_in[7];
  float* out = (float*)d_out;

  char* ws = (char*)d_ws;
  const size_t MB = 1ull << 20;
  bf16_t* xb     = (bf16_t*)(ws + 0);         // 16 MB
  bf16_t* WqT    = (bf16_t*)(ws + 16 * MB);   //  8 MB
  bf16_t* WdownT = (bf16_t*)(ws + 24 * MB);   //  2 MB
  bf16_t* WkupT  = (bf16_t*)(ws + 26 * MB);   //  2 MB
  bf16_t* WvupT  = (bf16_t*)(ws + 28 * MB);   //  2 MB
  bf16_t* WoT    = (bf16_t*)(ws + 30 * MB);   //  8 MB
  float*  Qf     = (float*) (ws + 38 * MB);   // 32 MB (dead after RoPE)
  float*  ckvf   = (float*) (ws + 70 * MB);   //  8 MB
  bf16_t* ckvb   = (bf16_t*)(ws + 78 * MB);   //  4 MB
  float*  Kf     = (float*) (ws + 82 * MB);   // 32 MB
  float*  Vf     = (float*) (ws + 114 * MB);  // 32 MB
  bf16_t* Qb     = (bf16_t*)(ws + 146 * MB);  // 16 MB
  bf16_t* Kb     = (bf16_t*)(ws + 162 * MB);  // 16 MB
  bf16_t* Vtb    = (bf16_t*)(ws + 38 * MB);   // 16 MB, reuses Qf
  bf16_t* attnb  = (bf16_t*)(ws + 54 * MB);   // 16 MB, reuses Qf tail

  auto gemm = [&](const bf16_t* A, const bf16_t* Bt, float* C, int M, int N, int K) {
    dim3 grid(N / 128, M / 128);
    gemm_bt<<<grid, 256, 0, stream>>>(A, Bt, C, M, N, K);
  };

  // 1. activations + weights to bf16 (weights transposed)
  cvt_bf16<<<(MTOT * DD) / 256, 256, 0, stream>>>(x, xb, MTOT * DD);
  transpose_cvt<<<dim3(DD / 32, DD / 32),  256, 0, stream>>>(Wq,    WqT,    DD,  DD);
  transpose_cvt<<<dim3(DD / 32, LATD / 32),256, 0, stream>>>(Wdown, WdownT, DD,  LATD);
  transpose_cvt<<<dim3(LATD / 32, DD / 32),256, 0, stream>>>(Wkup,  WkupT,  LATD, DD);
  transpose_cvt<<<dim3(LATD / 32, DD / 32),256, 0, stream>>>(Wvup,  WvupT,  LATD, DD);
  transpose_cvt<<<dim3(DD / 32, DD / 32),  256, 0, stream>>>(Wo,    WoT,    DD,  DD);

  // 2. projections
  gemm(xb, WqT, Qf, MTOT, DD, DD);            // Q  = x @ Wq
  gemm(xb, WdownT, ckvf, MTOT, LATD, DD);     // c_kv = x @ Wdown
  cvt_bf16<<<(MTOT * LATD) / 256, 256, 0, stream>>>(ckvf, ckvb, MTOT * LATD);
  gemm(ckvb, WkupT, Kf, MTOT, DD, LATD);      // K = c_kv @ Wkup
  gemm(ckvb, WvupT, Vf, MTOT, DD, LATD);      // V = c_kv @ Wvup

  // 3. RoPE on Q,K; V transpose to [B,H,HD,S]
  rope_cvt<<<(BB * SS * HH * (HDIM / 2)) / 256, 256, 0, stream>>>(Qf, freqs, Qb);
  rope_cvt<<<(BB * SS * HH * (HDIM / 2)) / 256, 256, 0, stream>>>(Kf, freqs, Kb);
  v_transpose<<<(BB * SS * HH * HDIM) / 256, 256, 0, stream>>>(Vf, Vtb);

  // 4. flash attention (one wave per b,h,q-tile)
  mla_attn<<<(BB * HH * (SS / 16)) / 8, 256, 0, stream>>>(Qb, Kb, Vtb, attnb);

  // 5. output projection -> fp32 d_out
  gemm(attnb, WoT, out, MTOT, DD, DD);
}
